// AttentionContextExtractor_83597243449803
// MI455X (gfx1250) — matrix operand order repaired and checked by reading, not compiled, not run on hardware
//
#include <hip/hip_runtime.h>

// Causal MHA forward for MI455X (gfx1250, wave32, WMMA).
// All matrix math via v_wmma_f32_16x16x32_bf16 with f32 accumulation.
// Round 3: restore the round-1 unrolled GEMM k-loop (best codegen: s_clause'd
// loads + 8 WMMAs per step); keep flash's loads-before-consumers ordering.
// Workspace: ~88 MB (bf16 copies of x, Wq..Wo, Q, K, V^T, context).

typedef unsigned short u16;
typedef __attribute__((ext_vector_type(16))) __bf16 v16bf;
typedef __attribute__((ext_vector_type(8)))  float  v8f;

#define NB   4      // batch
#define LL   2048   // sequence length
#define DM   1024   // d_model
#define NH   16     // heads
#define DH   64     // head dim
#define BLR  (NB * LL)   // 8192 rows

__device__ __forceinline__ u16 f2bf(float f) {
  unsigned u = __float_as_uint(f);
  u += 0x7FFFu + ((u >> 16) & 1u);          // round-to-nearest-even
  return (u16)(u >> 16);
}

// A-matrix fragment (16x32 bf16): lane half h holds contiguous K = 16h..16h+15
// caller passes p = row_base + k0 + half*16
__device__ __forceinline__ v16bf load_a_frag(const u16* p) {
  union { uint4 q[2]; v16bf v; } u;
  u.q[0] = *reinterpret_cast<const uint4*>(p);
  u.q[1] = *reinterpret_cast<const uint4*>(p + 8);
  return u.v;
}
// B-matrix fragment (32x16 bf16): lane half h holds contiguous K = 16h..16h+15
// caller passes p = col_base + k0 + half*16
__device__ __forceinline__ v16bf load_b_frag(const u16* p) {
  union { uint4 q[2]; v16bf v; } u;
  u.q[0] = *reinterpret_cast<const uint4*>(p);
  u.q[1] = *reinterpret_cast<const uint4*>(p + 8);
  return u.v;
}

__device__ __forceinline__ v8f wmma_bf16(v16bf a, v16bf b, v8f c) {
  return __builtin_amdgcn_wmma_f32_16x16x32_bf16(
      /*neg_a=*/false, a, /*neg_b=*/false, b,
      /*c_mod=*/(short)0, c, /*reuse_a=*/false, /*reuse_b=*/false);
}

// ---------------- fp32 -> bf16 conversion (4 elems / thread) ----------------
__global__ __launch_bounds__(256) void f32_to_bf16_kernel(
    const float* __restrict__ in, u16* __restrict__ out, int n4) {
  int i = blockIdx.x * blockDim.x + threadIdx.x;
  if (i >= n4) return;
  float4 v = reinterpret_cast<const float4*>(in)[i];
  uint2 o;
  o.x = (unsigned)f2bf(v.x) | ((unsigned)f2bf(v.y) << 16);
  o.y = (unsigned)f2bf(v.z) | ((unsigned)f2bf(v.w) << 16);
  reinterpret_cast<uint2*>(out)[i] = o;
}

// ---------------- GEMM: C[M,N] = A[M,K] * W[N,K]^T -------------------------
// Wave tile 32x64 (2x4 accumulators). Block = 8 waves -> 64x256 tile.
// MODE 0: store f32 to Cf[M,N] (+optional bf16 Cb)
// MODE 1: store bf16 head-major [b,h,l,DH]  (Q / K)
// MODE 2: store bf16 head-major transposed [b,h,DH,l]  (V^T)
template <int MODE>
__global__ __launch_bounds__(256) void gemm_bf16_kernel(
    const u16* __restrict__ A, const u16* __restrict__ W,
    float* __restrict__ Cf, u16* __restrict__ Cb,
    int M, int N, int K, float scale) {
  const int lane  = threadIdx.x & 31;
  const int wave  = threadIdx.x >> 5;
  const int lhalf = lane >> 4;
  const int l15   = lane & 15;
  const int m0 = blockIdx.y * 64  + (wave >> 2) * 32;
  const int n0 = blockIdx.x * 256 + (wave & 3) * 64;

  v8f acc[2][4] = {};
  const u16* a0 = A + (size_t)(m0 + l15) * K + lhalf * 16;
  const u16* a1 = A + (size_t)(m0 + 16 + l15) * K + lhalf * 16;

#pragma unroll 4
  for (int k = 0; k < K; k += 32) {
    v16bf af[2];
    af[0] = load_a_frag(a0 + k);
    af[1] = load_a_frag(a1 + k);
    v16bf bf[4];
#pragma unroll
    for (int j = 0; j < 4; ++j)
      bf[j] = load_b_frag(W + (size_t)(n0 + j * 16 + l15) * K + k + lhalf * 16);
#pragma unroll
    for (int mi = 0; mi < 2; ++mi)
#pragma unroll
      for (int nj = 0; nj < 4; ++nj)
        acc[mi][nj] = wmma_bf16(af[mi], bf[nj], acc[mi][nj]);
  }

#pragma unroll
  for (int mi = 0; mi < 2; ++mi)
#pragma unroll
    for (int nj = 0; nj < 4; ++nj)
#pragma unroll
      for (int r = 0; r < 8; ++r) {
        const int m = m0 + mi * 16 + lhalf * 8 + r;   // C row (lane-half layout)
        const int n = n0 + nj * 16 + l15;             // C col
        const float v = acc[mi][nj][r] * scale;
        if (MODE == 0) {
          Cf[(size_t)m * N + n] = v;
          if (Cb) Cb[(size_t)m * N + n] = f2bf(v);
        } else {
          const int bb = m >> 11, ll = m & (LL - 1);
          const int hh = n >> 6,  dd = n & (DH - 1);
          if (MODE == 1)
            Cb[(((size_t)(bb * NH + hh)) * LL + ll) * DH + dd] = f2bf(v);
          else
            Cb[(((size_t)(bb * NH + hh)) * DH + dd) * LL + ll] = f2bf(v);
        }
      }
}

// ---------------- Flash attention (causal, online softmax) -----------------
// One wave per 16-row query tile; 32-key blocks; Q pre-scaled by 1/sqrt(DH).
// All K and V fragments for a block are issued before any consumer: the V
// loads hide behind the score WMMAs and the softmax VALU section.
__global__ __launch_bounds__(256) void flash_attn_kernel(
    const u16* __restrict__ Qb, const u16* __restrict__ Kb,
    const u16* __restrict__ Vt, float* __restrict__ ctxf,
    u16* __restrict__ ctxb) {
  __shared__ __align__(16) u16 Plds[8][16][32];   // per-wave P staging (C->A layout)
  const int lane  = threadIdx.x & 31;
  const int wave  = threadIdx.x >> 5;
  const int lhalf = lane >> 4;
  const int l15   = lane & 15;
  const int bh = blockIdx.y;                 // b*NH + h
  const int qt = blockIdx.x * 8 + wave;      // query tile
  const int q0 = qt * 16;
  const u16* Q  = Qb + (size_t)bh * LL * DH;
  const u16* Kp = Kb + (size_t)bh * LL * DH;
  const u16* Vp = Vt + (size_t)bh * DH * LL;

  v16bf aq[2];
#pragma unroll
  for (int t = 0; t < 2; ++t)
    aq[t] = load_a_frag(Q + (size_t)(q0 + l15) * DH + t * 32 + lhalf * 16);

  v8f o[4] = {};
  float mI[8], lI[8];
#pragma unroll
  for (int r = 0; r < 8; ++r) { mI[r] = -__builtin_inff(); lI[r] = 0.f; }

  const int nkb = (q0 + 47) >> 5;            // 32-key blocks covering keys 0..q0+15
  for (int kb = 0; kb < nkb; ++kb) {
    const int kk0 = kb * 32;
    if (kk0 + 32 < LL) {                     // pull next block toward L2/L0
      __builtin_prefetch(Kp + (size_t)(kk0 + 32 + l15) * DH, 0, 0);
      __builtin_prefetch(Vp + (size_t)l15 * LL + kk0 + 32, 0, 0);
    }

    // Issue ALL loads for this block first, consume later.
    v16bf kf[2][2];                          // [key group][k-step]
#pragma unroll
    for (int g = 0; g < 2; ++g) {
      const u16* krow = Kp + (size_t)(kk0 + g * 16 + l15) * DH;
      kf[g][0] = load_b_frag(krow + lhalf * 16);
      kf[g][1] = load_b_frag(krow + 32 + lhalf * 16);
    }
    v16bf vf[4];
#pragma unroll
    for (int t = 0; t < 4; ++t)
      vf[t] = load_b_frag(Vp + (size_t)(t * 16 + l15) * LL + kk0 + lhalf * 16);

    // S = Q * K^T : two 16-key column groups
    v8f s[2] = {};
#pragma unroll
    for (int g = 0; g < 2; ++g) {
      s[g] = wmma_bf16(aq[0], kf[g][0], s[g]);
      s[g] = wmma_bf16(aq[1], kf[g][1], s[g]);
    }

    // online softmax; row m = lhalf*8 + r lives across the 16 lanes of a half
#pragma unroll
    for (int r = 0; r < 8; ++r) {
      const int qg = q0 + lhalf * 8 + r;
      float s0 = (kk0 + l15      <= qg) ? s[0][r] : -__builtin_inff();
      float s1 = (kk0 + 16 + l15 <= qg) ? s[1][r] : -__builtin_inff();
      float rm = fmaxf(s0, s1);
#pragma unroll
      for (int off = 8; off; off >>= 1) rm = fmaxf(rm, __shfl_xor(rm, off, 32));
      const float mNew  = fmaxf(mI[r], rm);
      const float alpha = __expf(mI[r] - mNew);
      const float p0 = __expf(s0 - mNew);    // masked lanes: exp(-inf)=0
      const float p1 = __expf(s1 - mNew);
      float ps = p0 + p1;
#pragma unroll
      for (int off = 8; off; off >>= 1) ps += __shfl_xor(ps, off, 32);
      lI[r] = lI[r] * alpha + ps;
      mI[r] = mNew;
#pragma unroll
      for (int t = 0; t < 4; ++t) o[t][r] *= alpha;
      Plds[wave][lhalf * 8 + r][l15]      = f2bf(p0);
      Plds[wave][lhalf * 8 + r][16 + l15] = f2bf(p1);
    }
    __builtin_amdgcn_wave_barrier();         // keep LDS writes before reads (DS in-order per wave)

    // O += P * V  (V fragments already in flight / resident)
    v16bf pa = load_a_frag(&Plds[wave][l15][0] + lhalf * 16);
#pragma unroll
    for (int t = 0; t < 4; ++t)
      o[t] = wmma_bf16(pa, vf[t], o[t]);
  }

  const int bb = bh >> 4, hh = bh & 15;
#pragma unroll
  for (int r = 0; r < 8; ++r) {
    const int l = q0 + lhalf * 8 + r;
    const float inv = 1.0f / lI[r];
#pragma unroll
    for (int t = 0; t < 4; ++t) {
      const int d = t * 16 + l15;
      const float v = o[t][r] * inv;
      const size_t idx = ((size_t)(bb * LL + l)) * DM + hh * DH + d;
      ctxf[idx] = v;                          // fp32 context (2nd tuple output)
      ctxb[idx] = f2bf(v);                    // bf16 copy for the Wo GEMM
    }
  }
}

extern "C" void kernel_launch(void* const* d_in, const int* in_sizes, int n_in,
                              void* d_out, int out_size, void* d_ws, size_t ws_size,
                              hipStream_t stream) {
  (void)in_sizes; (void)n_in; (void)out_size; (void)ws_size;
  const float* x  = (const float*)d_in[0];
  const float* Wq = (const float*)d_in[1];
  const float* Wk = (const float*)d_in[2];
  const float* Wv = (const float*)d_in[3];
  const float* Wo = (const float*)d_in[4];
  // d_in[5] = attn_mask (tril) — causality is applied analytically in-kernel.

  float* out  = (float*)d_out;                    // [B,L,D] output
  float* ctxf = out + (size_t)BLR * DM;           // [B,L,D] context

  char* ws = (char*)d_ws;
  size_t off = 0;
  auto take = [&](size_t bytes) -> char* {
    char* p = ws + off;
    off += (bytes + 255) & ~(size_t)255;
    return p;
  };
  u16* xb  = (u16*)take((size_t)BLR * DM * 2);    // bf16 x
  u16* wqb = (u16*)take((size_t)DM * DM * 2);
  u16* wkb = (u16*)take((size_t)DM * DM * 2);
  u16* wvb = (u16*)take((size_t)DM * DM * 2);
  u16* wob = (u16*)take((size_t)DM * DM * 2);
  u16* Qb  = (u16*)take((size_t)BLR * DM * 2);    // [b,h,l,DH], pre-scaled
  u16* Kb  = (u16*)take((size_t)BLR * DM * 2);    // [b,h,l,DH]
  u16* Vt  = (u16*)take((size_t)BLR * DM * 2);    // [b,h,DH,l]
  u16* cxb = (u16*)take((size_t)BLR * DM * 2);    // bf16 context

  // 1) down-convert activations and weights
  {
    const int n4 = BLR * DM / 4;
    f32_to_bf16_kernel<<<(n4 + 255) / 256, 256, 0, stream>>>(x, xb, n4);
    const int w4 = DM * DM / 4;
    f32_to_bf16_kernel<<<(w4 + 255) / 256, 256, 0, stream>>>(Wq, wqb, w4);
    f32_to_bf16_kernel<<<(w4 + 255) / 256, 256, 0, stream>>>(Wk, wkb, w4);
    f32_to_bf16_kernel<<<(w4 + 255) / 256, 256, 0, stream>>>(Wv, wvb, w4);
    f32_to_bf16_kernel<<<(w4 + 255) / 256, 256, 0, stream>>>(Wo, wob, w4);
  }

  // 2) projections (scale 1/sqrt(DH)=0.125 folded into Q; V stored transposed)
  dim3 gg(DM / 256, BLR / 64), bb(256);
  gemm_bf16_kernel<1><<<gg, bb, 0, stream>>>(xb, wqb, nullptr, Qb, BLR, DM, DM, 0.125f);
  gemm_bf16_kernel<1><<<gg, bb, 0, stream>>>(xb, wkb, nullptr, Kb, BLR, DM, DM, 1.0f);
  gemm_bf16_kernel<2><<<gg, bb, 0, stream>>>(xb, wvb, nullptr, Vt, BLR, DM, DM, 1.0f);

  // 3) causal flash attention -> context (fp32 out + bf16 scratch)
  flash_attn_kernel<<<dim3(LL / 128, NB * NH), 256, 0, stream>>>(Qb, Kb, Vt, ctxf, cxb);

  // 4) output projection -> first half of d_out
  gemm_bf16_kernel<0><<<gg, bb, 0, stream>>>(cxb, wob, out, nullptr, BLR, DM, DM, 1.0f);
}
